// spatialDiscriminator_80633716015377
// MI455X (gfx1250) — matrix operand order repaired and checked by reading, not compiled, not run on hardware
//
#include <hip/hip_runtime.h>

#define N_AQI 35
#define N_MEO 18
#define NN    53
#define NP    64
#define RS    66   // padded LDS row stride in f16 elements (33 dwords, odd -> conflict-free)
#define F     16
#define ALPHA 0.2f
#define BT_TOTAL 1024

typedef __attribute__((ext_vector_type(16))) _Float16 v16h;
typedef __attribute__((ext_vector_type(8)))  float    v8f;

union Frag { unsigned u[8]; v16h h; };

// weight-pack offsets inside wl[]
#define O_WT   0     // W_trans (6)
#define O_BT   6     // b_trans (6)
#define O_WXA  12    // W_xa (6x16)
#define O_WXM  108   // W_xm (3x16)
#define O_WUA  156   // W_ua (6x16)
#define O_WUM  252   // W_um (3x16)
#define O_AAA  300   // a_aa (33)
#define O_AAM  333   // a_am (33)
#define O_AMA  366   // a_ma (33)
#define O_AMM  399   // a_mm (33)
#define O_WO   432   // W_out (32)
#define O_BO   464   // b_out (1)

__global__ __launch_bounds__(32)
void spatial_disc_fused(const float* __restrict__ out_aqi,
                        const float* __restrict__ out_meo,
                        const float* __restrict__ adj,
                        const float* __restrict__ adj_norm,
                        const float* __restrict__ W_trans,
                        const float* __restrict__ b_trans,
                        const float* __restrict__ W_xa,
                        const float* __restrict__ W_xm,
                        const float* __restrict__ W_ua,
                        const float* __restrict__ W_um,
                        const float* __restrict__ a_aa,
                        const float* __restrict__ a_am,
                        const float* __restrict__ a_ma,
                        const float* __restrict__ a_mm,
                        const float* __restrict__ W_out,
                        const float* __restrict__ b_out,
                        float* __restrict__ out_sscore,
                        float* __restrict__ out_smetric)
{
    __shared__ __align__(16) _Float16 attL[NP * RS];  // exp(e - rowmax), f16, zero-padded
    __shared__ __align__(16) _Float16 hatT[F * RS];   // heter_attri transposed [f][node], f16
    __shared__ float s0[NP], s1[NP], d0[NP], d1[NP], rscale[NP];
    __shared__ float wl[468];

    const int lane = threadIdx.x;
    const int bt   = blockIdx.x;

    // ---- load all small weights into LDS (cooperative) ----
    for (int i = lane; i < 6;  i += 32) { wl[O_WT + i] = W_trans[i]; wl[O_BT + i] = b_trans[i]; }
    for (int i = lane; i < 96; i += 32) { wl[O_WXA + i] = W_xa[i];   wl[O_WUA + i] = W_ua[i]; }
    for (int i = lane; i < 48; i += 32) { wl[O_WXM + i] = W_xm[i];   wl[O_WUM + i] = W_um[i]; }
    for (int i = lane; i < 33; i += 32) {
        wl[O_AAA + i] = a_aa[i]; wl[O_AAM + i] = a_am[i];
        wl[O_AMA + i] = a_ma[i]; wl[O_AMM + i] = a_mm[i];
    }
    wl[O_WO + lane] = W_out[lane];
    if (lane == 0) wl[O_BO] = b_out[0];

    // ---- zero padded LDS tiles ----
    {
        unsigned* za = (unsigned*)attL;               // NP*RS/2 = 2112 dwords
        for (int i = lane; i < (NP * RS) / 2; i += 32) za[i] = 0u;
        unsigned* zh = (unsigned*)hatT;               // F*RS/2 = 528 dwords
        for (int i = lane; i < (F * RS) / 2; i += 32) zh[i] = 0u;
    }
    __syncthreads();

    // ---- per-node features: heter_attri (-> hatT f16) and attention s/d terms ----
    for (int n = lane; n < NP; n += 32) {
        float ha[F];
        float sv0 = 0.f, sv1 = 0.f, dv0 = 0.f, dv1 = 0.f;
        if (n < N_AQI) {
            float x = out_aqi[bt * N_AQI + n];
            float h[F];
            #pragma unroll
            for (int f = 0; f < F; f++) { ha[f] = 0.f; h[f] = 0.f; }
            #pragma unroll
            for (int k = 0; k < 6; k++) {
                float a6 = fmaf(x, wl[O_WT + k], wl[O_BT + k]);
                #pragma unroll
                for (int f = 0; f < F; f++) {
                    ha[f] = fmaf(a6, wl[O_WXA + k * F + f], ha[f]);
                    h[f]  = fmaf(a6, wl[O_WUA + k * F + f], h[f]);
                }
            }
            #pragma unroll
            for (int f = 0; f < F; f++) {
                sv0 = fmaf(h[f], wl[O_AAA + f],      sv0);   // src term vs AQI dsts
                sv1 = fmaf(h[f], wl[O_AAM + f],      sv1);   // src term vs MEO dsts
                dv0 = fmaf(h[f], wl[O_AAA + 16 + f], dv0);   // dst term under AQI srcs
                dv1 = fmaf(h[f], wl[O_AMA + 16 + f], dv1);   // dst term under MEO srcs
            }
        } else if (n < NN) {
            int m = n - N_AQI;
            float m0 = out_meo[(bt * N_MEO + m) * 3 + 0];
            float m1 = out_meo[(bt * N_MEO + m) * 3 + 1];
            float m2 = out_meo[(bt * N_MEO + m) * 3 + 2];
            float h[F];
            #pragma unroll
            for (int f = 0; f < F; f++) {
                ha[f] = fmaf(m0, wl[O_WXM + 0 * F + f],
                        fmaf(m1, wl[O_WXM + 1 * F + f], m2 * wl[O_WXM + 2 * F + f]));
                h[f]  = fmaf(m0, wl[O_WUM + 0 * F + f],
                        fmaf(m1, wl[O_WUM + 1 * F + f], m2 * wl[O_WUM + 2 * F + f]));
            }
            #pragma unroll
            for (int f = 0; f < F; f++) {
                sv0 = fmaf(h[f], wl[O_AMA + f],      sv0);
                sv1 = fmaf(h[f], wl[O_AMM + f],      sv1);
                dv0 = fmaf(h[f], wl[O_AAM + 16 + f], dv0);
                dv1 = fmaf(h[f], wl[O_AMM + 16 + f], dv1);
            }
        } else {
            #pragma unroll
            for (int f = 0; f < F; f++) ha[f] = 0.f;
        }
        s0[n] = sv0; s1[n] = sv1; d0[n] = dv0; d1[n] = dv1;
        #pragma unroll
        for (int f = 0; f < F; f++) hatT[f * RS + n] = (_Float16)ha[f];
    }
    __syncthreads();

    // ---- edge scores + masked softmax (unnormalized exp -> attL, 1/sum -> rscale) ----
    for (int i = lane; i < NP; i += 32) {
        if (i >= NN) { rscale[i] = 0.f; continue; }
        const bool  srcA = (i < N_AQI);
        const float s0i = s0[i], s1i = s1[i];
        const float cJA = srcA ? wl[O_AAA + 32] : wl[O_AMA + 32];
        const float cJM = srcA ? wl[O_AAM + 32] : wl[O_AMM + 32];
        const float* arow  = adj      + i * NN;
        const float* anrow = adj_norm + i * NN;

        float vals[NN];
        float vmax = -1e12f;
        #pragma unroll
        for (int j = 0; j < NN; j++) {
            float an = anrow[j];
            float e  = (j < N_AQI ? fmaf(an, cJA, s0i) : fmaf(an, cJM, s1i))
                       + (srcA ? d0[j] : d1[j]);
            e = (e > 0.f) ? e : (ALPHA * e);                 // leaky relu
            float v = (arow[j] > 0.f) ? e : -1e12f;          // adjacency mask
            vals[j] = v;
            vmax = fmaxf(vmax, v);
        }
        float sum = 0.f;
        #pragma unroll
        for (int j = 0; j < NN; j++) {
            float ex = __expf(vals[j] - vmax);               // masked -> exp(-1e12-max) = 0
            sum += ex;
            attL[i * RS + j] = (_Float16)ex;
        }
        rscale[i] = 1.f / sum;
    }
    __syncthreads();

    // ---- heter_out = att @ heter_attri via V_WMMA_F32_16X16X32_F16 ----
    const int half = lane >> 4;       // 0: lanes 0-15, 1: lanes 16-31
    const int mlo  = lane & 15;       // A-matrix M index / C-matrix N index

    // B fragments (32x16 f16, K-contiguous in hatT), reused across all 4 M tiles
    Frag bf[2];
    #pragma unroll
    for (int kc = 0; kc < 2; kc++) {
        #pragma unroll
        for (int v = 0; v < 8; v++) {
            int K = kc * 32 + half * 16 + 2 * v;
            bf[kc].u[v] = *(const unsigned*)&hatT[mlo * RS + K];
        }
    }

    float sumA = 0.f, sumM = 0.f;     // per-lane partial column sums (col = mlo)
    #pragma unroll
    for (int t = 0; t < 4; t++) {
        const int M = t * 16 + mlo;   // A-matrix row for this lane
        v8f acc = {};
        #pragma unroll
        for (int kc = 0; kc < 2; kc++) {
            Frag af;
            #pragma unroll
            for (int v = 0; v < 8; v++) {
                int k0 = 2 * v + 16 * half;
                af.u[v] = *(const unsigned*)&attL[M * RS + kc * 32 + k0];
            }
            acc = __builtin_amdgcn_wmma_f32_16x16x32_f16(
                      false, af.h, false, bf[kc].h, (short)0, acc, false, false);
        }
        #pragma unroll
        for (int r = 0; r < 8; r++) {
            int row = t * 16 + r + 8 * half;          // C/D layout: VGPR r, lane half
            float vout = acc[r] * rscale[row];        // softmax normalization folded here
            if (row < N_AQI) sumA += vout;
            else             sumM += vout;            // rows >= 53 carry rscale = 0
        }
    }

    // combine the two half-wave row subsets (same column in both halves)
    sumA += __shfl_xor(sumA, 16);
    sumM += __shfl_xor(sumM, 16);

    // smetric: lanes 0-15 -> sigmoid(mean over AQI rows), 16-31 -> sigmoid(mean over MEO rows)
    float mean = (lane < 16) ? sumA * (1.f / N_AQI) : sumM * (1.f / N_MEO);
    float sm = 1.f / (1.f + __expf(-mean));
    out_smetric[bt * 32 + lane] = sm;

    // sscore = sigmoid(smetric . W_out + b_out): wave32 reduction
    float r = sm * wl[O_WO + lane];
    #pragma unroll
    for (int off = 16; off > 0; off >>= 1) r += __shfl_xor(r, off);
    if (lane == 0) out_sscore[bt] = 1.f / (1.f + __expf(-(r + wl[O_BO])));
}

extern "C" void kernel_launch(void* const* d_in, const int* in_sizes, int n_in,
                              void* d_out, int out_size, void* d_ws, size_t ws_size,
                              hipStream_t stream) {
    (void)n_in; (void)out_size; (void)d_ws; (void)ws_size;
    const float* out_aqi  = (const float*)d_in[0];
    const float* out_meo  = (const float*)d_in[1];
    const float* adj      = (const float*)d_in[2];
    const float* adj_norm = (const float*)d_in[3];
    const float* W_trans  = (const float*)d_in[4];
    const float* b_trans  = (const float*)d_in[5];
    const float* W_xa     = (const float*)d_in[6];
    const float* W_xm     = (const float*)d_in[7];
    const float* W_ua     = (const float*)d_in[8];
    const float* W_um     = (const float*)d_in[9];
    const float* a_aa     = (const float*)d_in[10];
    const float* a_am     = (const float*)d_in[11];
    const float* a_ma     = (const float*)d_in[12];
    const float* a_mm     = (const float*)d_in[13];
    const float* W_out    = (const float*)d_in[14];
    const float* b_out    = (const float*)d_in[15];

    const int BT = in_sizes[0] / N_AQI;   // 1024
    float* sscore  = (float*)d_out;        // (BT,1)
    float* smetric = (float*)d_out + BT;   // (BT,32)

    spatial_disc_fused<<<BT, 32, 0, stream>>>(
        out_aqi, out_meo, adj, adj_norm, W_trans, b_trans, W_xa, W_xm,
        W_ua, W_um, a_aa, a_am, a_ma, a_mm, W_out, b_out, sscore, smetric);
}
